// _EquiformerBlock_32229434589410
// MI455X (gfx1250) — matrix operand order repaired and checked by reading, not compiled, not run on hardware
//
#include <hip/hip_runtime.h>
#include <hip/hip_bf16.h>

// EquiformerBlock for MI455X (gfx1250, wave32, WMMA).
// All GEMMs use v_wmma_f32_16x16x32_f16 (f16 operands, fp32 accumulate).
// The dominant bias-MLP (262144x64 -> 768 -> 12, ~31 GFLOP) is fully fused in
// LDS so its 768MB intermediate never hits HBM; everything else is ~memory-
// bound (~75MB total -> ~3us at 23.3 TB/s HBM).
// Round 2: fast-rcp sigmoid/softmax/LN (kill v_div_scale Newton chains seen in
// the round-1 asm), global_prefetch_b8 of the next K-tile in the GEMM loop.

typedef _Float16 f16;
typedef __attribute__((ext_vector_type(16))) _Float16 v16h;
typedef __attribute__((ext_vector_type(8)))  float    v8f;

#define Bq 4
#define Nq 256
#define Dq 768
#define Hq 12
#define HDq 64
#define TOK (Bq * Nq)          // 1024 token rows
#define RROWS (Bq * Nq * Nq)   // 262144 bias-MLP rows

__device__ __forceinline__ v8f wmma_f16(v16h a, v16h b, v8f c) {
  return __builtin_amdgcn_wmma_f32_16x16x32_f16(false, a, false, b, (short)0, c,
                                                false, false);
}

// K index of component c of the 16-bit A fragment (ISA 7.12.2, 16x32 A):
// VGPR0-3 hold K = (lanehalf*8)+0..7, VGPR4-7 hold K = 16+(lanehalf*8)+0..7
__device__ __forceinline__ int a_k(int lane, int c) {
  int kb = (lane >> 4) * 8;
  return (c < 8) ? (kb + c) : (16 + kb + (c - 8));
}

// fast sigmoid: one v_exp_f32 + one v_rcp_f32 (no IEEE div Newton chain)
__device__ __forceinline__ float sigm(float x) {
  return __builtin_amdgcn_rcpf(1.f + __expf(-x));
}

// ---------------------------------------------------------------- converters
__global__ void cvt_f32_f16(const float* __restrict__ s, f16* __restrict__ d, int n) {
  int i = blockIdx.x * 256 + threadIdx.x;
  if (i < n) d[i] = (f16)s[i];
}

// rb_w2 [768,12] -> padded [768,16] f16 (cols 12..15 = 0)
__global__ void cvt_pad_rbw2(const float* __restrict__ s, f16* __restrict__ d) {
  int i = blockIdx.x * 256 + threadIdx.x;
  if (i < 768 * 16) {
    int k = i >> 4, n = i & 15;
    d[i] = (f16)(n < 12 ? s[k * 12 + n] : 0.f);
  }
}

// ---------------------------------------------------------------- layernorm
__global__ void __launch_bounds__(256)
ln_kernel(const float* __restrict__ x, const float* __restrict__ g,
          const float* __restrict__ b, f16* __restrict__ out) {
  __shared__ float rs[256], rs2[256];
  int row = blockIdx.x, t = threadIdx.x;
  const float* xr = x + (size_t)row * Dq;
  float s = 0.f, s2 = 0.f;
  for (int i = t; i < Dq; i += 256) { float v = xr[i]; s += v; s2 += v * v; }
  rs[t] = s; rs2[t] = s2;
  __syncthreads();
  for (int off = 128; off > 0; off >>= 1) {
    if (t < off) { rs[t] += rs[t + off]; rs2[t] += rs2[t + off]; }
    __syncthreads();
  }
  const float invD = 1.f / (float)Dq;          // constant multiply, no div
  float mu = rs[0] * invD;
  float var = rs2[0] * invD - mu * mu;
  float rstd = rsqrtf(var + 1e-5f);
  for (int i = t; i < Dq; i += 256)
    out[(size_t)row * Dq + i] = (f16)((xr[i] - mu) * rstd * g[i] + b[i]);
}

// ---------------------------------------------------------------- generic GEMM
// C[M,N] = A_h[M,K] @ B_h[K,N] + bias, epilogue EPI:
//  0: +bias -> f16    1: silu -> f16    2: sigmoid -> f32
//  3: +bias +resid -> f32               4: gelu(exact) -> f16
template <int EPI>
__device__ __forceinline__ void epi_store(float v, size_t idx,
                                          const float* __restrict__ resid,
                                          float* __restrict__ Cf,
                                          f16* __restrict__ Ch) {
  if (EPI == 0) Ch[idx] = (f16)v;
  else if (EPI == 1) Ch[idx] = (f16)(v * sigm(v));
  else if (EPI == 2) Cf[idx] = sigm(v);
  else if (EPI == 3) Cf[idx] = v + resid[idx];
  else if (EPI == 4) Ch[idx] = (f16)(0.5f * v * (1.f + erff(v * 0.70710678f)));
}

template <int EPI>
__global__ void __launch_bounds__(256)
gemm_kernel(const f16* __restrict__ A, const f16* __restrict__ Bm,
            const float* __restrict__ bias, const float* __restrict__ resid,
            float* __restrict__ Cf, f16* __restrict__ Ch, int M, int N, int K) {
  __shared__ f16 As[64 * 32];
  __shared__ f16 Bs[32 * 64];
  int t = threadIdx.x, lane = t & 31, w = t >> 5;
  int wr = w >> 1, wc = w & 1;               // wave tile: rows wr*16, cols wc*32
  int m0 = blockIdx.y * 64, n0 = blockIdx.x * 64;
  int ar = t >> 2, ac = (t & 3) * 8;         // A: 8 halfs / thread
  int br = t >> 3, bc = (t & 7) * 8;         // B: 8 halfs / thread
  v8f acc0 = {}, acc1 = {};
  for (int k0 = 0; k0 < K; k0 += 32) {
    __syncthreads();
    *(uint4*)&As[ar * 32 + ac] = *(const uint4*)&A[(size_t)(m0 + ar) * K + k0 + ac];
    *(uint4*)&Bs[br * 64 + bc] = *(const uint4*)&Bm[(size_t)(k0 + br) * N + n0 + bc];
    if (k0 + 32 < K) {                       // warm next K-tile: global_prefetch_b8
      __builtin_prefetch(&A[(size_t)(m0 + ar) * K + k0 + 32 + ac], 0, 0);
      __builtin_prefetch(&Bm[(size_t)(k0 + 32 + br) * N + n0 + bc], 0, 0);
    }
    __syncthreads();
    v16h a, b0, b1;
    int m = wr * 16 + (lane & 15);
#pragma unroll
    for (int c = 0; c < 16; ++c) a[c] = As[m * 32 + a_k(lane, c)];
#pragma unroll
    for (int c = 0; c < 16; ++c) {
      b0[c] = Bs[a_k(lane, c) * 64 + wc * 32 + (lane & 15)];
      b1[c] = Bs[a_k(lane, c) * 64 + wc * 32 + 16 + (lane & 15)];
    }
    acc0 = wmma_f16(a, b0, acc0);
    acc1 = wmma_f16(a, b1, acc1);
  }
#pragma unroll
  for (int r = 0; r < 8; ++r) {
    int mm = m0 + wr * 16 + r + 8 * (lane >> 4);
    int nnA = n0 + wc * 32 + (lane & 15);
    int nnB = nnA + 16;
    epi_store<EPI>(acc0[r] + bias[nnA], (size_t)mm * N + nnA, resid, Cf, Ch);
    epi_store<EPI>(acc1[r] + bias[nnB], (size_t)mm * N + nnB, resid, Cf, Ch);
  }
}

// ------------------------------------------------- fused distance-bias MLP
// 16 rows/WG: silu(dist[16,64] @ w1[64,768] + b1) @ w2[768,16] + b2
__global__ void __launch_bounds__(256)
bias_mlp_kernel(const float* __restrict__ dist, const f16* __restrict__ w1,
                const float* __restrict__ b1, const f16* __restrict__ w2p,
                const float* __restrict__ b2, float* __restrict__ bias_pad) {
  __shared__ f16 Ds[16 * 64];
  __shared__ f16 Hs[16 * 768];
  __shared__ float Ps[8 * 16 * 16];
  int t = threadIdx.x, lane = t & 31, w = t >> 5;
  size_t r0 = (size_t)blockIdx.x * 16;
  for (int i = t; i < 16 * 64; i += 256) Ds[i] = (f16)dist[r0 * 64 + i];
  __syncthreads();
  // layer 1: each of 8 waves produces 6 of the 48 N-tiles
  for (int s = 0; s < 6; ++s) {
    int n0 = (w * 6 + s) * 16;
    v8f acc = {};
    for (int kk = 0; kk < 2; ++kk) {
      int k0 = kk * 32;
      v16h a, b;
      int m = lane & 15;
#pragma unroll
      for (int c = 0; c < 16; ++c) a[c] = Ds[m * 64 + k0 + a_k(lane, c)];
#pragma unroll
      for (int c = 0; c < 16; ++c) b[c] = w1[(k0 + a_k(lane, c)) * 768 + n0 + (lane & 15)];
      acc = wmma_f16(a, b, acc);
    }
#pragma unroll
    for (int r = 0; r < 8; ++r) {
      int mm = r + 8 * (lane >> 4);
      int nn = n0 + (lane & 15);
      float v = acc[r] + b1[nn];
      Hs[mm * 768 + nn] = (f16)(v * sigm(v));
    }
  }
  __syncthreads();
  // layer 2: split K=768 over 8 waves (3 WMMA steps each), reduce in LDS
  {
    v8f acc = {};
    for (int kk = 0; kk < 3; ++kk) {
      int k0 = w * 96 + kk * 32;
      v16h a, b;
      int m = lane & 15;
#pragma unroll
      for (int c = 0; c < 16; ++c) a[c] = Hs[m * 768 + k0 + a_k(lane, c)];
#pragma unroll
      for (int c = 0; c < 16; ++c) b[c] = w2p[(k0 + a_k(lane, c)) * 16 + (lane & 15)];
      acc = wmma_f16(a, b, acc);
    }
#pragma unroll
    for (int r = 0; r < 8; ++r)
      Ps[(w * 16 + r + 8 * (lane >> 4)) * 16 + (lane & 15)] = acc[r];
  }
  __syncthreads();
  {
    int mm = t >> 4, nn = t & 15;
    float sum = 0.f;
#pragma unroll
    for (int ww = 0; ww < 8; ++ww) sum += Ps[(ww * 16 + mm) * 16 + nn];
    sum += (nn < 12) ? b2[nn] : 0.f;
    bias_pad[(r0 + mm) * 16 + nn] = sum;
  }
}

// ------------------------------------------------------------- attention
// One block per (b, h, 16-row i-tile): logits + bias + mask -> softmax ->
// attn@V, fused with sigmoid gate, stores f16 input of the out-projection.
__global__ void __launch_bounds__(128)
attn_kernel(const f16* __restrict__ qkvh, const float* __restrict__ bias_pad,
            const unsigned char* __restrict__ mask,
            const float* __restrict__ gate_sig, f16* __restrict__ ctxgate) {
  __shared__ float Ls[16 * 256];
  __shared__ f16 Ah[16 * 256];
  __shared__ float red[16 * 8];
  int t = threadIdx.x, lane = t & 31, w = t >> 5;   // 4 waves
  int bid = blockIdx.x;
  int it = bid & 15, h = (bid >> 4) % Hq, b = bid / (16 * Hq);
  int i0 = it * 16;
  const int LDQ = 3 * Dq, KOFF = Dq, VOFF = 2 * Dq;
  // logits: wave w handles j-tiles w*4 .. w*4+3
  for (int s = 0; s < 4; ++s) {
    int j0 = (w * 4 + s) * 16;
    v8f acc = {};
    for (int kk = 0; kk < 2; ++kk) {
      int d0 = kk * 32;
      v16h a, bf;
      int m = lane & 15;
#pragma unroll
      for (int c = 0; c < 16; ++c)
        a[c] = qkvh[(size_t)(b * Nq + i0 + m) * LDQ + h * HDq + d0 + a_k(lane, c)];
#pragma unroll
      for (int c = 0; c < 16; ++c)
        bf[c] = qkvh[(size_t)(b * Nq + j0 + (lane & 15)) * LDQ + KOFF + h * HDq + d0 + a_k(lane, c)];
      acc = wmma_f16(a, bf, acc);
    }
#pragma unroll
    for (int r = 0; r < 8; ++r) {
      int mi = r + 8 * (lane >> 4);
      int j = j0 + (lane & 15);
      size_t e = (size_t)(b * Nq + i0 + mi) * Nq + j;
      float v = acc[r] * 0.125f + bias_pad[e * 16 + h];      // 1/sqrt(64)
      Ls[mi * 256 + j] = mask[e] ? v : -1e30f;
    }
  }
  __syncthreads();
  // softmax: 8 threads per row, 32 columns each
  int row = t >> 3, sub = t & 7;
  float lmax = -1e30f;
  for (int c = sub * 32; c < sub * 32 + 32; ++c) lmax = fmaxf(lmax, Ls[row * 256 + c]);
  red[row * 8 + sub] = lmax;
  __syncthreads();
  float rmax = -1e30f;
#pragma unroll
  for (int k2 = 0; k2 < 8; ++k2) rmax = fmaxf(rmax, red[row * 8 + k2]);
  __syncthreads();
  float lsum = 0.f;
  for (int c = sub * 32; c < sub * 32 + 32; ++c) {
    float e = __expf(Ls[row * 256 + c] - rmax);
    Ls[row * 256 + c] = e;
    lsum += e;
  }
  red[row * 8 + sub] = lsum;
  __syncthreads();
  float rsum = 0.f;
#pragma unroll
  for (int k2 = 0; k2 < 8; ++k2) rsum += red[row * 8 + k2];
  float inv = __builtin_amdgcn_rcpf(rsum);                   // fast v_rcp_f32
  for (int c = sub * 32; c < sub * 32 + 32; ++c)
    Ah[row * 256 + c] = (f16)(Ls[row * 256 + c] * inv);
  __syncthreads();
  // context: wave w owns head-dim n-tile w*16, K = 256 -> 8 WMMA steps
  {
    int n0 = w * 16;
    v8f acc = {};
    for (int kk = 0; kk < 8; ++kk) {
      int j0 = kk * 32;
      v16h a, bf;
      int m = lane & 15;
#pragma unroll
      for (int c = 0; c < 16; ++c) a[c] = Ah[m * 256 + j0 + a_k(lane, c)];
#pragma unroll
      for (int c = 0; c < 16; ++c)
        bf[c] = qkvh[(size_t)(b * Nq + j0 + a_k(lane, c)) * LDQ + VOFF + h * HDq + n0 + (lane & 15)];
      acc = wmma_f16(a, bf, acc);
    }
#pragma unroll
    for (int r = 0; r < 8; ++r) {
      int i = i0 + r + 8 * (lane >> 4);
      int col = h * HDq + n0 + (lane & 15);
      size_t idx = (size_t)(b * Nq + i) * Dq + col;
      ctxgate[idx] = (f16)(acc[r] * gate_sig[idx]);
    }
  }
}

// ---------------------------------------------------------------- launcher
extern "C" void kernel_launch(void* const* d_in, const int* in_sizes, int n_in,
                              void* d_out, int out_size, void* d_ws, size_t ws_size,
                              hipStream_t stream) {
  (void)in_sizes; (void)n_in; (void)out_size; (void)ws_size;
  const float* x      = (const float*)d_in[0];
  const float* dist   = (const float*)d_in[1];
  const unsigned char* emask = (const unsigned char*)d_in[2];   // jax bool = 1B
  const float* qkv_w  = (const float*)d_in[3];
  const float* qkv_b  = (const float*)d_in[4];
  const float* out_w  = (const float*)d_in[5];
  const float* out_b  = (const float*)d_in[6];
  const float* rb_w1  = (const float*)d_in[7];
  const float* rb_b1  = (const float*)d_in[8];
  const float* rb_w2  = (const float*)d_in[9];
  const float* rb_b2  = (const float*)d_in[10];
  const float* gate_w1 = (const float*)d_in[11];
  const float* gate_b1 = (const float*)d_in[12];
  const float* gate_w2 = (const float*)d_in[13];
  const float* gate_b2 = (const float*)d_in[14];
  const float* ff_w1  = (const float*)d_in[15];
  const float* ff_b1  = (const float*)d_in[16];
  const float* ff_w2  = (const float*)d_in[17];
  const float* ff_b2  = (const float*)d_in[18];
  const float* ln1_g  = (const float*)d_in[19];
  const float* ln1_b  = (const float*)d_in[20];
  const float* ln2_g  = (const float*)d_in[21];
  const float* ln2_b  = (const float*)d_in[22];
  float* out = (float*)d_out;

  char* p = (char*)d_ws;
  auto alloc = [&](size_t bytes) {
    void* r = (void*)p;
    p += (bytes + 255) & ~(size_t)255;
    return r;
  };
  f16* qkv_w_h  = (f16*)alloc((size_t)Dq * 3 * Dq * 2);
  f16* out_w_h  = (f16*)alloc((size_t)Dq * Dq * 2);
  f16* rb_w1_h  = (f16*)alloc((size_t)HDq * Dq * 2);
  f16* rb_w2p_h = (f16*)alloc((size_t)Dq * 16 * 2);
  f16* gate_w1_h = (f16*)alloc((size_t)Dq * Dq * 2);
  f16* gate_w2_h = (f16*)alloc((size_t)Dq * Dq * 2);
  f16* ff_w1_h  = (f16*)alloc((size_t)Dq * 2 * Dq * 2);
  f16* ff_w2_h  = (f16*)alloc((size_t)2 * Dq * Dq * 2);
  f16* xn_h     = (f16*)alloc((size_t)TOK * Dq * 2);
  f16* qkv_h    = (f16*)alloc((size_t)TOK * 3 * Dq * 2);
  f16* g1_h     = (f16*)alloc((size_t)TOK * Dq * 2);
  float* gate_sig = (float*)alloc((size_t)TOK * Dq * 4);
  float* bias_pad = (float*)alloc((size_t)RROWS * 16 * 4);
  f16* ctxgate_h = (f16*)alloc((size_t)TOK * Dq * 2);
  float* x2     = (float*)alloc((size_t)TOK * Dq * 4);
  f16* y_h      = (f16*)alloc((size_t)TOK * Dq * 2);
  f16* ff1_h    = (f16*)alloc((size_t)TOK * 2 * Dq * 2);

  auto cvt = [&](const float* s, f16* d, int n) {
    cvt_f32_f16<<<(n + 255) / 256, 256, 0, stream>>>(s, d, n);
  };
  cvt(qkv_w,  qkv_w_h,  Dq * 3 * Dq);
  cvt(out_w,  out_w_h,  Dq * Dq);
  cvt(rb_w1,  rb_w1_h,  HDq * Dq);
  cvt_pad_rbw2<<<(768 * 16 + 255) / 256, 256, 0, stream>>>(rb_w2, rb_w2p_h);
  cvt(gate_w1, gate_w1_h, Dq * Dq);
  cvt(gate_w2, gate_w2_h, Dq * Dq);
  cvt(ff_w1,  ff_w1_h,  Dq * 2 * Dq);
  cvt(ff_w2,  ff_w2_h,  2 * Dq * Dq);

  // LN1
  ln_kernel<<<TOK, 256, 0, stream>>>(x, ln1_g, ln1_b, xn_h);
  // qkv = xn @ qkv_w + b   (f16 out)
  gemm_kernel<0><<<dim3(3 * Dq / 64, TOK / 64), 256, 0, stream>>>(
      xn_h, qkv_w_h, qkv_b, nullptr, nullptr, qkv_h, TOK, 3 * Dq, Dq);
  // gate path: silu(xn@W1+b1) -> sigmoid(.@W2+b2)
  gemm_kernel<1><<<dim3(Dq / 64, TOK / 64), 256, 0, stream>>>(
      xn_h, gate_w1_h, gate_b1, nullptr, nullptr, g1_h, TOK, Dq, Dq);
  gemm_kernel<2><<<dim3(Dq / 64, TOK / 64), 256, 0, stream>>>(
      g1_h, gate_w2_h, gate_b2, nullptr, gate_sig, nullptr, TOK, Dq, Dq);
  // fused distance-bias MLP (the 31-GFLOP hot spot)
  bias_mlp_kernel<<<RROWS / 16, 256, 0, stream>>>(
      dist, rb_w1_h, rb_b1, rb_w2p_h, rb_b2, bias_pad);
  // attention + gate fusion
  attn_kernel<<<Bq * Hq * (Nq / 16), 128, 0, stream>>>(
      qkv_h, bias_pad, emask, gate_sig, ctxgate_h);
  // out projection + residual -> x2 (fp32)
  gemm_kernel<3><<<dim3(Dq / 64, TOK / 64), 256, 0, stream>>>(
      ctxgate_h, out_w_h, out_b, x, x2, nullptr, TOK, Dq, Dq);
  // LN2 -> y
  ln_kernel<<<TOK, 256, 0, stream>>>(x2, ln2_g, ln2_b, y_h);
  // FF: gelu(y@W1+b1) @ W2 + b2 + x2 -> out
  gemm_kernel<4><<<dim3(2 * Dq / 64, TOK / 64), 256, 0, stream>>>(
      y_h, ff_w1_h, ff_b1, nullptr, nullptr, ff1_h, TOK, 2 * Dq, Dq);
  gemm_kernel<3><<<dim3(Dq / 64, TOK / 64), 256, 0, stream>>>(
      ff1_h, ff_w2_h, ff_b2, x2, out, nullptr, TOK, Dq, 2 * Dq);
}
